// SparseINN_90056874262873
// MI455X (gfx1250) — compile-verified, matrix operand order checked by reading, not run on hardware
//
#include <hip/hip_runtime.h>

// ---------------------------------------------------------------------------
// SparseINN attention pipeline for MI455X (gfx1250, wave32, WMMA bf16 + TDM)
//
// Shapes: B=1024, N=64 neurons, D=1024, H=16 heads, HD=64, TOPK=16, NSTAT=8.
//
// ws layout (bytes):
//   [0)        Wqkv bf16 [3*1024][1024]
//   [WO_OFF)   Wo   bf16 [1024][1024]
//   [STAT_OFF) stat bf16 [H][8][64]
//   [Q_OFF)    Q    bf16 [B][H][64][64]
//   [K_OFF)    K    bf16 [B][H][64][64]
//   [VT_OFF)   Vt   bf16 [B][H][64][64]      (V transposed: [dh][n])
//   [AO_OFF)   AOut bf16 [B][64][1024]
//   [X_OFF)    Xbf  bf16 [B][64][1024]
// total ~650 MiB
// ---------------------------------------------------------------------------

typedef __attribute__((ext_vector_type(16))) __bf16 v16bf;
typedef __attribute__((ext_vector_type(8)))  __bf16 v8bf;
typedef __attribute__((ext_vector_type(8)))  float  v8f;
typedef __attribute__((ext_vector_type(4)))  unsigned int v4u;
typedef __attribute__((ext_vector_type(8)))  int v8i;
typedef __attribute__((ext_vector_type(4)))  int v4i;

#define NB      1024
#define NNEUR   64
#define DMODEL  1024
#define NHEAD   16
#define HDIM    64
#define TOPK    16
#define NSTAT   8

// LDS slab row stride: 1024 bf16 + TDM pad of 2 dwords every 256 dwords
// -> 516 dwords = 1032 elements per row.  516 % 64 == 4 banks, so the 16
// per-lane ds_read_b128 fragment fetches hit distinct banks.
#define XS_STRIDE 1032

static const size_t WQKV_OFF = 0;
static const size_t WO_OFF   = (size_t)3 * 1024 * 1024 * 2;           //  6 MiB
static const size_t STAT_OFF = WO_OFF + (size_t)1024 * 1024 * 2;      //  8 MiB
static const size_t Q_OFF    = STAT_OFF + 131072;
static const size_t QKV_SZ   = (size_t)NB * NHEAD * NNEUR * HDIM * 2; // 128 MiB
static const size_t K_OFF    = Q_OFF + QKV_SZ;
static const size_t VT_OFF   = K_OFF + QKV_SZ;
static const size_t AO_OFF   = VT_OFF + QKV_SZ;
static const size_t AO_SZ    = (size_t)NB * NNEUR * DMODEL * 2;       // 128 MiB
static const size_t X_OFF    = AO_OFF + AO_SZ;

// ---- fragment helper -------------------------------------------------------
// 16-bit A/B fragment layout (ISA 7.12.2): per lane, two contiguous 16-byte
// chunks at +0 and +32 bytes (elements k..k+7 and k+16..k+23).
__device__ __forceinline__ v16bf frag_bf16(const __bf16* p) {
    v8bf lo = *(const v8bf*)p;
    v8bf hi = *(const v8bf*)(p + 16);
    v16bf r;
#pragma unroll
    for (int i = 0; i < 8; ++i) { r[i] = lo[i]; r[i + 8] = hi[i]; }
    return r;
}

// Index into the TDM-padded LDS slab: element k of row n.
// Rows are 512 dwords of data with a 2-dword pad after each 256 dwords, so
// elements >= 512 sit 4 elements later.  Fragment chunks (k ≡ 0/8 mod 32,
// len 8, pair at +16) never straddle the pad.
__device__ __forceinline__ int xs_idx(int n, int k) {
    return n * XS_STRIDE + k + (k >= 512 ? 4 : 0);
}

// ---- Tensor Data Mover: load one [64 x 1024] bf16 slab into LDS -----------
// D# per ISA §8: group0 = {count, lds_addr, global_addr, type=2},
// group1 = {data_size=2B, pad cfg, tensor 1024x64, tile 1024x64, stride 1024}.
// pad_interval=7 (256 dwords), pad_amount=1 (2 dwords) -> row stride 516 dw.
// This toolchain exposes the 6-arg builtin (extra int32x8 group before cpol).
__device__ __forceinline__ void tdm_load_slab(const __bf16* gsrc,
                                              __bf16* lds_dst) {
    unsigned long long ga = (unsigned long long)(uintptr_t)gsrc;
    unsigned lds_addr = (unsigned)(uintptr_t)lds_dst;  // low 32b = LDS offset
    v4u g0;
    g0[0] = 1u;                                        // count = 1
    g0[1] = lds_addr;
    g0[2] = (unsigned)ga;
    g0[3] = ((unsigned)(ga >> 32) & 0x01FFFFFFu) | (2u << 30);   // type=2
    v8i g1;
    g1[0] = (int)((1u << 16)        // data_size = 2 bytes
                | (1u << 20)        // pad_enable
                | (7u << 22)        // pad_interval = 256 dwords
                | (1u << 25));      // pad_amount   = 2 dwords
    g1[1] = (int)(1024u << 16);     // tensor_dim0[15:0] = 1024
    g1[2] = (int)(64u << 16);       // tensor_dim1[15:0] = 64
    g1[3] = (int)(1024u << 16);     // tile_dim0 = 1024
    g1[4] = 64;                     // tile_dim1 = 64
    g1[5] = 1024;                   // tensor_dim0_stride = 1024
    g1[6] = 0;
    g1[7] = 0;
    v4i z4 = {0, 0, 0, 0};
    v8i z8 = {0, 0, 0, 0, 0, 0, 0, 0};
#if __has_builtin(__builtin_amdgcn_tensor_load_to_lds)
    __builtin_amdgcn_tensor_load_to_lds(g0, g1, z4, z4, z8, 0);
    __builtin_amdgcn_s_wait_tensorcnt(0);
#else
    // fallback: cooperative wave copy
    const unsigned* s = (const unsigned*)gsrc;
    unsigned* d = (unsigned*)lds_dst;
    for (int i = (int)(threadIdx.x & 31); i < 64 * 512; i += 32) {
        int n = i >> 9, k2 = i & 511;
        d[n * (XS_STRIDE / 2) + k2 + (k2 >= 256 ? 2 : 0)] = s[i];
    }
#endif
}

// ---- kernel 1a: f32 -> bf16 (weights) -------------------------------------
__global__ void sinn_cvt_f32_bf16(const float* __restrict__ src,
                                  __bf16* __restrict__ dst, int n) {
    int i = blockIdx.x * blockDim.x + threadIdx.x;
    int stride = gridDim.x * blockDim.x;
    for (; i < n; i += stride) dst[i] = (__bf16)src[i];
}

// ---- kernel 1b: x f32 -> bf16, vectorized (8 elements / thread) -----------
__global__ void sinn_cvt_x(const float* __restrict__ src,
                           __bf16* __restrict__ dst, long long n8) {
    long long t = (long long)blockIdx.x * blockDim.x + threadIdx.x;
    if (t >= n8) return;
    const float4* s = (const float4*)(src + t * 8);
    float4 f0 = s[0], f1 = s[1];
    float v[8] = {f0.x, f0.y, f0.z, f0.w, f1.x, f1.y, f1.z, f1.w};
    v8bf o;
#pragma unroll
    for (int i = 0; i < 8; ++i) o[i] = (__bf16)v[i];
    *(v8bf*)(dst + t * 8) = o;
}

// ---- kernel 1c: static keys, per-head layout ------------------------------
__global__ void sinn_cvt_static(const float* __restrict__ sk,
                                __bf16* __restrict__ stat) {
    int i = blockIdx.x * blockDim.x + threadIdx.x;   // 8*1024 elements
    if (i >= NSTAT * DMODEL) return;
    int m = i / DMODEL, c = i % DMODEL;
    int h = c >> 6, dh = c & 63;
    stat[(h * NSTAT + m) * HDIM + dh] = (__bf16)sk[i];
}

// ---- kernel 2: fused QKV projection ---------------------------------------
// grid = (48, B), block = 128 (4 waves). Whole x_b slab comes in via TDM;
// wave w owns 16 fused-output columns c0 = bx*64 + w*16 and all 64 rows.
__global__ void __launch_bounds__(128)
sinn_qkv(const __bf16* __restrict__ xbf,
         const __bf16* __restrict__ wqkv,
         const float* __restrict__ qb, const float* __restrict__ kb,
         const float* __restrict__ vb,
         __bf16* __restrict__ qo, __bf16* __restrict__ ko,
         __bf16* __restrict__ vto) {
    __shared__ __bf16 xs[NNEUR * XS_STRIDE];         // 132 KB padded slab

    const int tid  = threadIdx.x;
    const int w    = tid >> 5;
    const int lane = tid & 31;
    const int lm   = lane & 15;
    const int half = lane >> 4;
    const int b    = blockIdx.y;
    const int c0   = blockIdx.x * 64 + w * 16;       // uniform per wave

    if (w == 0) tdm_load_slab(xbf + (size_t)b * NNEUR * DMODEL, xs);
    __syncthreads();

    v8f acc[4] = {};
#pragma unroll 4
    for (int k0 = 0; k0 < DMODEL; k0 += 32) {
        const int kk = k0 + 8 * half;
        const __bf16* pb = wqkv + (size_t)(c0 + lm) * DMODEL + kk;
        v16bf bfrag = frag_bf16(pb);
#pragma unroll
        for (int mt = 0; mt < 4; ++mt) {
            v16bf afrag = frag_bf16(xs + xs_idx(mt * 16 + lm, kk));
            acc[mt] = __builtin_amdgcn_wmma_f32_16x16x32_bf16(
                false, afrag, false, bfrag, (short)0, acc[mt], false, false);
        }
    }

    const int region = c0 >> 10;                     // 0=Q,1=K,2=V (uniform)
    const int ch = (c0 & 1023) + lm;
    const int h  = ch >> 6, dh = ch & 63;
    const float* bias = (region == 0) ? qb : (region == 1) ? kb : vb;
    const float bv = bias[ch];

    if (region < 2) {
        __bf16* dst = (region == 0) ? qo : ko;       // [b][h][n][dh]
        size_t base = (((size_t)b * NHEAD + h) * NNEUR) * HDIM + dh;
#pragma unroll
        for (int mt = 0; mt < 4; ++mt)
#pragma unroll
            for (int r = 0; r < 8; ++r) {
                int n = mt * 16 + half * 8 + r;
                dst[base + (size_t)n * HDIM] = (__bf16)(acc[mt][r] + bv);
            }
    } else {                                         // Vt: [b][h][dh][n]
        size_t rowb = (((size_t)b * NHEAD + h) * HDIM + dh) * NNEUR;
#pragma unroll
        for (int mt = 0; mt < 4; ++mt) {
            v8bf pk;
#pragma unroll
            for (int r = 0; r < 8; ++r) pk[r] = (__bf16)(acc[mt][r] + bv);
            *(v8bf*)(vto + rowb + mt * 16 + half * 8) = pk;
        }
    }
}

// ---- kernel 3: sparse attention per (b, h) --------------------------------
// grid = (H, B), block = 128 (4 waves). Wave w owns score rows [w*16,w*16+16).
__global__ void __launch_bounds__(128)
sinn_attn(const __bf16* __restrict__ qi, const __bf16* __restrict__ ki,
          const __bf16* __restrict__ vti, const __bf16* __restrict__ stat,
          __bf16* __restrict__ ao) {
    __shared__ __bf16 kbuf[NNEUR * HDIM];   //  8 KB effective keys
    __shared__ float  sbuf[NNEUR * NNEUR];  // 16 KB scores
    __shared__ __bf16 pbuf[NNEUR * NNEUR];  //  8 KB probs

    const int tid  = threadIdx.x;
    const int w    = tid >> 5;
    const int lane = tid & 31;
    const int lm   = lane & 15;
    const int half = lane >> 4;
    const int h    = blockIdx.x;
    const int b    = blockIdx.y;
    const size_t bh = (size_t)b * NHEAD + h;

    // build effective K (static rows 0..7, projected rows 8..63) in LDS
    {
        const unsigned* ksrc = (const unsigned*)(ki + bh * NNEUR * HDIM);
        const unsigned* ssrc = (const unsigned*)(stat + (size_t)h * NSTAT * HDIM);
        unsigned* kd = (unsigned*)kbuf;
        for (int i = tid; i < NNEUR * (HDIM / 2); i += 128) {
            int m = i >> 5, d2 = i & 31;
            kd[i] = (m < NSTAT) ? ssrc[m * 32 + d2] : ksrc[i];
        }
    }
    __syncthreads();

    // S = Q K'^T / 8
    const __bf16* qrow0 = qi + (bh * NNEUR + w * 16 + lm) * HDIM;
    v8f s[4] = {};
#pragma unroll
    for (int k0 = 0; k0 < HDIM; k0 += 32) {
        v16bf afrag = frag_bf16(qrow0 + k0 + 8 * half);
#pragma unroll
        for (int ct = 0; ct < 4; ++ct) {
            v16bf bfrag = frag_bf16(kbuf + (ct * 16 + lm) * HDIM + k0 + 8 * half);
            s[ct] = __builtin_amdgcn_wmma_f32_16x16x32_bf16(
                false, afrag, false, bfrag, (short)0, s[ct], false, false);
        }
    }
#pragma unroll
    for (int ct = 0; ct < 4; ++ct)
#pragma unroll
        for (int r = 0; r < 8; ++r)
            sbuf[(w * 16 + half * 8 + r) * NNEUR + ct * 16 + lm] =
                s[ct][r] * 0.125f;
    __syncthreads();

    // per-row exact top-16 (with multiplicity) threshold + masked softmax
    if (tid < NNEUR) {
        float* row = sbuf + tid * NNEUR;
        float cur = __builtin_inff();
        int need = TOPK;
        for (int it = 0; it < TOPK && need > 0; ++it) {
            float m = -__builtin_inff();
            for (int j = 0; j < NNEUR; ++j) {
                float v = row[j];
                if (v < cur) m = fmaxf(m, v);
            }
            int c = 0;
            for (int j = 0; j < NNEUR; ++j) c += (row[j] == m);
            need -= c;
            cur = m;
        }
        const float kth = cur;
        float rmax = -__builtin_inff();
        for (int j = 0; j < NNEUR; ++j) rmax = fmaxf(rmax, row[j]);
        float sum = 0.f;
        for (int j = 0; j < NNEUR; ++j) {
            float v = row[j];
            float e = (v >= kth) ? expf(v - rmax) : 0.f;
            row[j] = e;
            sum += e;
        }
        float inv = 1.f / sum;
        for (int j = 0; j < NNEUR; ++j)
            pbuf[tid * NNEUR + j] = (__bf16)(row[j] * inv);
    }
    __syncthreads();

    // O = P V   (B columns = Vt rows, contiguous)
    const __bf16* prow0 = pbuf + (w * 16 + lm) * NNEUR;
    const __bf16* vtb   = vti + bh * HDIM * NNEUR;
    v8f o[4] = {};
#pragma unroll
    for (int k0 = 0; k0 < NNEUR; k0 += 32) {
        v16bf afrag = frag_bf16(prow0 + k0 + 8 * half);
#pragma unroll
        for (int ct = 0; ct < 4; ++ct) {
            v16bf bfrag = frag_bf16(vtb + (size_t)(ct * 16 + lm) * NNEUR +
                                    k0 + 8 * half);
            o[ct] = __builtin_amdgcn_wmma_f32_16x16x32_bf16(
                false, afrag, false, bfrag, (short)0, o[ct], false, false);
        }
    }
    // attn-out row major [b][n][D], column = h*64 + dh
#pragma unroll
    for (int ct = 0; ct < 4; ++ct) {
        int dh = ct * 16 + lm;
#pragma unroll
        for (int r = 0; r < 8; ++r) {
            int n = w * 16 + half * 8 + r;
            ao[((size_t)b * NNEUR + n) * DMODEL + h * HDIM + dh] =
                (__bf16)o[ct][r];
        }
    }
}

// ---- kernel 4: output projection (f32 result) -----------------------------
// grid = (16, B), block = 128 (4 waves). Attn-out slab comes in via TDM.
__global__ void __launch_bounds__(128)
sinn_oproj(const __bf16* __restrict__ ain, const __bf16* __restrict__ wo,
           const float* __restrict__ ob, float* __restrict__ out) {
    __shared__ __bf16 xs[NNEUR * XS_STRIDE];         // 132 KB padded slab

    const int tid  = threadIdx.x;
    const int w    = tid >> 5;
    const int lane = tid & 31;
    const int lm   = lane & 15;
    const int half = lane >> 4;
    const int b    = blockIdx.y;
    const int c0   = blockIdx.x * 64 + w * 16;

    if (w == 0) tdm_load_slab(ain + (size_t)b * NNEUR * DMODEL, xs);
    __syncthreads();

    v8f acc[4] = {};
#pragma unroll 4
    for (int k0 = 0; k0 < DMODEL; k0 += 32) {
        const int kk = k0 + 8 * half;
        v16bf bfrag = frag_bf16(wo + (size_t)(c0 + lm) * DMODEL + kk);
#pragma unroll
        for (int mt = 0; mt < 4; ++mt) {
            v16bf afrag = frag_bf16(xs + xs_idx(mt * 16 + lm, kk));
            acc[mt] = __builtin_amdgcn_wmma_f32_16x16x32_bf16(
                false, afrag, false, bfrag, (short)0, acc[mt], false, false);
        }
    }
    const float bv = ob[c0 + lm];
    float* ob_out = out + (size_t)b * NNEUR * DMODEL + c0 + lm;
#pragma unroll
    for (int mt = 0; mt < 4; ++mt)
#pragma unroll
        for (int r = 0; r < 8; ++r) {
            int n = mt * 16 + half * 8 + r;
            ob_out[(size_t)n * DMODEL] = acc[mt][r] + bv;   // coalesced rows
        }
}

// ---------------------------------------------------------------------------
extern "C" void kernel_launch(void* const* d_in, const int* in_sizes, int n_in,
                              void* d_out, int out_size, void* d_ws,
                              size_t ws_size, hipStream_t stream) {
    const float* x    = (const float*)d_in[0];
    const float* qW   = (const float*)d_in[1];
    const float* qb   = (const float*)d_in[2];
    const float* kW   = (const float*)d_in[3];
    const float* kb   = (const float*)d_in[4];
    const float* vW   = (const float*)d_in[5];
    const float* vb   = (const float*)d_in[6];
    const float* oW   = (const float*)d_in[7];
    const float* ob   = (const float*)d_in[8];
    const float* stk  = (const float*)d_in[9];
    float* out        = (float*)d_out;

    char* ws = (char*)d_ws;
    __bf16* wqkv = (__bf16*)(ws + WQKV_OFF);
    __bf16* wob  = (__bf16*)(ws + WO_OFF);
    __bf16* stat = (__bf16*)(ws + STAT_OFF);
    __bf16* qws  = (__bf16*)(ws + Q_OFF);
    __bf16* kws  = (__bf16*)(ws + K_OFF);
    __bf16* vtws = (__bf16*)(ws + VT_OFF);
    __bf16* aows = (__bf16*)(ws + AO_OFF);
    __bf16* xbf  = (__bf16*)(ws + X_OFF);

    const int NW = DMODEL * DMODEL;
    sinn_cvt_f32_bf16<<<2048, 256, 0, stream>>>(qW, wqkv, NW);
    sinn_cvt_f32_bf16<<<2048, 256, 0, stream>>>(kW, wqkv + (size_t)NW, NW);
    sinn_cvt_f32_bf16<<<2048, 256, 0, stream>>>(vW, wqkv + (size_t)2 * NW, NW);
    sinn_cvt_f32_bf16<<<2048, 256, 0, stream>>>(oW, wob, NW);
    sinn_cvt_static<<<(NSTAT * DMODEL + 255) / 256, 256, 0, stream>>>(stk, stat);

    const long long n8 = (long long)NB * NNEUR * DMODEL / 8;
    sinn_cvt_x<<<(unsigned)((n8 + 255) / 256), 256, 0, stream>>>(x, xbf, n8);

    sinn_qkv<<<dim3(48, NB), 128, 0, stream>>>(xbf, wqkv, qb, kb, vb,
                                               qws, kws, vtws);
    sinn_attn<<<dim3(NHEAD, NB), 128, 0, stream>>>(qws, kws, vtws, stat, aows);
    sinn_oproj<<<dim3(16, NB), 128, 0, stream>>>(aows, wob, ob, out);
}
